// Encoder_59004260712867
// MI455X (gfx1250) — compile-verified, matrix operand order checked by reading, not run on hardware
//
#include <hip/hip_runtime.h>
#include <hip/hip_bf16.h>

// ---------------- problem geometry ----------------
static constexpr int BATCH = 64;
static constexpr int TLEN  = 256;
static constexpr int DSZ   = 1024;   // embed dim
static constexpr int HSZ   = 1024;   // hidden dim
static constexpr int LNUM  = 4;      // layers
static constexpr int NB    = 128;    // combined fwd+bwd batch rows
static constexpr int G4H   = 4 * HSZ;        // 4096 gate cols
static constexpr int KCAT  = DSZ + HSZ;      // 2048 concat K
static constexpr int KCHUNK = 64;            // k elems staged per TDM tile
static constexpr int NCHUNK = KCAT / KCHUNK; // 32

typedef __attribute__((ext_vector_type(16))) __bf16 v16bf;
typedef __attribute__((ext_vector_type(8)))  __bf16 v8bf;
typedef __attribute__((ext_vector_type(8)))  float  v8f;
typedef __attribute__((ext_vector_type(4)))  unsigned int u32x4;
typedef __attribute__((ext_vector_type(8)))  int i32x8;
typedef __attribute__((ext_vector_type(4)))  int i32x4;

// ---------------- WMMA fragment loader ----------------
// 16-bit A (16x32 MxK) / B (32x16 KxN) per-lane pattern (ISA 7.12.2):
//   row   = lane & 15 ;  kbase = (lane & 16) ? 8 : 0
//   elems 0..7 -> k0+kbase+0..7 ; elems 8..15 -> k0+kbase+16..23
// Works for global rows (stride KCAT) and LDS-staged rows (stride KCHUNK):
// two contiguous 16-byte loads per lane either way.
__device__ __forceinline__ v16bf load_frag(const __bf16* __restrict__ row, int k0, int lane) {
  const int kb = k0 + ((lane & 16) ? 8 : 0);
  v8bf lo = *reinterpret_cast<const v8bf*>(row + kb);
  v8bf hi = *reinterpret_cast<const v8bf*>(row + kb + 16);
  v16bf r;
#pragma unroll
  for (int i = 0; i < 8; ++i) { r[i] = lo[i]; r[i + 8] = hi[i]; }
  return r;
}

__device__ __forceinline__ float sigmoidf_(float x) { return 1.0f / (1.0f + __expf(-x)); }

// ---------------- TDM tile load (clang-23 6-arg builtin form) ----------------
// Stage a 16-row x 64-elem bf16 tile (global row stride KCAT elems) into
// LDS contiguously. Descriptor per CDNA5 ISA.
__device__ __forceinline__ void tdm_load_tile(const __bf16* gsrc, unsigned lds_off) {
  const unsigned long long ga = (unsigned long long)(size_t)gsrc;
  u32x4 g0;
  g0[0] = 1u;                                   // count=1, no gather
  g0[1] = lds_off;                              // lds_addr (bytes)
  g0[2] = (unsigned)(ga & 0xffffffffu);         // global_addr[31:0]
  g0[3] = (unsigned)((ga >> 32) & 0x01ffffffu)  // global_addr[56:32]
          | 0x80000000u;                        // type=2 ("image") at bits 127:126
  i32x8 g1;
  g1[0] = 0x00010000;                 // workgroup_mask=0 | data_size=1 (2B)
  g1[1] = (int)(2048u << 16);         // tensor_dim0 = 2048 (bits 79:48, low half)
  g1[2] = (int)(4096u << 16);         // td0 hi=0 | tensor_dim1 = 4096 (low 16)
  g1[3] = (int)(64u << 16);           // td1 hi=0 | tile_dim0 = 64
  g1[4] = 16;                         // tile_dim1 = 16 | tile_dim2 = 0
  g1[5] = 2048;                       // tensor_dim0_stride[31:0] = 2048 elems
  g1[6] = 0;                          // stride hi | td1_stride low
  g1[7] = 0;
  const i32x4 z4 = {0, 0, 0, 0};
  const i32x8 z8 = {0, 0, 0, 0, 0, 0, 0, 0};
  __builtin_amdgcn_tensor_load_to_lds(g0, g1, z4, z4, z8, 0);
}

// ---------------- prep kernels ----------------
__global__ void prep_wcat_kernel(const float* __restrict__ Wih, const float* __restrict__ Whh,
                                 __bf16* __restrict__ Wcat) {
  const size_t total = (size_t)LNUM * G4H * KCAT;
  for (size_t i = (size_t)blockIdx.x * blockDim.x + threadIdx.x; i < total;
       i += (size_t)gridDim.x * blockDim.x) {
    const size_t k  = i & (size_t)(KCAT - 1);
    const size_t ln = i >> 11;                       // l*4096 + n
    const float v = (k < (size_t)HSZ) ? Wih[ln * HSZ + k] : Whh[ln * HSZ + (k - HSZ)];
    Wcat[i] = (__bf16)v;
  }
}

__global__ void prep_bsum_kernel(const float* __restrict__ bih, const float* __restrict__ bhh,
                                 float* __restrict__ bsum) {
  const int total = LNUM * G4H;
  for (int i = blockIdx.x * blockDim.x + threadIdx.x; i < total; i += gridDim.x * blockDim.x)
    bsum[i] = bih[i] + bhh[i];
}

__global__ void prep_proj_kernel(const float* __restrict__ Wo, const float* __restrict__ Wh,
                                 const float* __restrict__ Wc, __bf16* __restrict__ WoB,
                                 __bf16* __restrict__ WhB, __bf16* __restrict__ WcB) {
  const size_t N = (size_t)HSZ * (2 * HSZ);
  for (size_t i = (size_t)blockIdx.x * blockDim.x + threadIdx.x; i < 3 * N;
       i += (size_t)gridDim.x * blockDim.x) {
    const size_t which = i / N, j = i - which * N;
    const float* s = (which == 0) ? Wo : (which == 1) ? Wh : Wc;
    __bf16*      d = (which == 0) ? WoB : (which == 1) ? WhB : WcB;
    d[j] = (__bf16)s[j];
  }
}

__global__ void embed_kernel(const int* __restrict__ tok, const float* __restrict__ emb,
                             __bf16* __restrict__ X) {
  const int bt = blockIdx.x;              // 0..B*T-1
  const int b = bt / TLEN, t = bt % TLEN;
  const int token = tok[b * TLEN + t];
  const float* src = emb + (size_t)token * DSZ;
  __bf16* dstF = X + ((size_t)t * NB + b) * DSZ;
  __bf16* dstR = dstF + (size_t)BATCH * DSZ;
  for (int j = threadIdx.x; j < DSZ; j += blockDim.x) {
    const __bf16 v = (__bf16)src[j];
    dstF[j] = v;
    dstR[j] = v;
  }
}

__global__ void zero_state_kernel(__bf16* __restrict__ hbf, float* __restrict__ c) {
  const int n = NB * HSZ;
  for (int i = blockIdx.x * blockDim.x + threadIdx.x; i < n; i += gridDim.x * blockDim.x) {
    hbf[i] = (__bf16)0.0f;
    c[i] = 0.0f;
  }
}

// ---------------- fused LSTM step ----------------
// gates[128,4096] = [Xsel | h] @ Wcat^T + bsum, elementwise LSTM, residual,
// bf16 repack. Block = 8 waves: jSel (2) x mQuad (4).
//   wave tile: 32 rows (2 m-tiles) x 16 hidden cols x 4 gates -> 8 accumulators.
// Weights staged via TDM: wave (jSel,mq) DMA-loads gate g=mq's 16x64 tile for
// its jSel into LDS (double buffered), consumed by all 4 m-quad waves.
__global__ void __launch_bounds__(256)
lstm_step_kernel(const __bf16* __restrict__ Xin, __bf16* __restrict__ Xout,
                 const __bf16* __restrict__ hIn, __bf16* __restrict__ hOut,
                 float* __restrict__ c, const __bf16* __restrict__ Wcat,
                 const float* __restrict__ bsum, __bf16* __restrict__ Hfin,
                 __bf16* __restrict__ Cfin, int s, int lastStep) {
  __shared__ __bf16 smem[2][2][4][16][KCHUNK];   // [buf][jSel][gate][row][k] = 32 KB

  const int lane = threadIdx.x & 31;
  const int w    = threadIdx.x >> 5;
  const int jSel = w & 1;
  const int mq   = w >> 1;                // 0..3, rows mq*32..mq*32+31
  const int jtile = blockIdx.x * 32 + jSel * 16;
  const int rsel  = lane & 15;

  // A-row pointers for the two 16-row m-tiles of this wave
  const __bf16* aX[2];
  const __bf16* aH[2];
#pragma unroll
  for (int p = 0; p < 2; ++p) {
    const int mrow = mq * 32 + p * 16 + rsel;
    const int trow = (mrow < BATCH) ? s : (TLEN - 1 - s);
    aX[p] = Xin + ((size_t)trow * NB + mrow) * HSZ;
    aH[p] = hIn + (size_t)mrow * HSZ;
  }

  // TDM source rows: this wave stages gate g=mq for its jSel
  const __bf16* wrow = Wcat + (size_t)(mq * HSZ + jtile) * KCAT;
  const unsigned ldsDst[2] = {
      (unsigned)(size_t)&smem[0][jSel][mq][0][0],
      (unsigned)(size_t)&smem[1][jSel][mq][0][0]};

  v8f acc[2][4];
#pragma unroll
  for (int p = 0; p < 2; ++p)
#pragma unroll
    for (int g = 0; g < 4; ++g) acc[p][g] = v8f{};

  tdm_load_tile(wrow, ldsDst[0]);                 // prologue: chunk 0

  for (int ch = 0; ch < NCHUNK; ++ch) {
    const int cur = ch & 1;
    if (ch + 1 < NCHUNK) {
      tdm_load_tile(wrow + (ch + 1) * KCHUNK, ldsDst[(ch + 1) & 1]);
      __builtin_amdgcn_s_wait_tensorcnt((short)1);   // chunk ch landed
    } else {
      __builtin_amdgcn_s_wait_tensorcnt((short)0);
    }
    __syncthreads();                                 // all waves' tiles visible

#pragma unroll
    for (int ks = 0; ks < KCHUNK; ks += 32) {
      const int kg = ch * KCHUNK + ks;
      v16bf a0, a1;
      if (kg < HSZ) {                                // uniform branch
        a0 = load_frag(aX[0], kg, lane);
        a1 = load_frag(aX[1], kg, lane);
      } else {
        a0 = load_frag(aH[0], kg - HSZ, lane);
        a1 = load_frag(aH[1], kg - HSZ, lane);
      }
#pragma unroll
      for (int g = 0; g < 4; ++g) {
        const v16bf bf_ = load_frag(&smem[cur][jSel][g][rsel][0], ks, lane);
        acc[0][g] = __builtin_amdgcn_wmma_f32_16x16x32_bf16(false, a0, false, bf_, (short)0,
                                                            acc[0][g], false, false);
        acc[1][g] = __builtin_amdgcn_wmma_f32_16x16x32_bf16(false, a1, false, bf_, (short)0,
                                                            acc[1][g], false, false);
      }
    }
    __syncthreads();                                 // safe to overwrite buf
  }

  // elementwise LSTM + residual
  const int ncol = jtile + rsel;
  const float bi = bsum[ncol];
  const float bf = bsum[HSZ + ncol];
  const float bg = bsum[2 * HSZ + ncol];
  const float bo = bsum[3 * HSZ + ncol];
#pragma unroll
  for (int p = 0; p < 2; ++p) {
    const int mbase = mq * 32 + p * 16 + ((lane & 16) ? 8 : 0);
#pragma unroll
    for (int vj = 0; vj < 8; ++vj) {
      const int m = mbase + vj;
      const float ig = sigmoidf_(acc[p][0][vj] + bi);
      const float fg = sigmoidf_(acc[p][1][vj] + bf);
      const float gg = tanhf(acc[p][2][vj] + bg);
      const float og = sigmoidf_(acc[p][3][vj] + bo);
      const size_t cidx = (size_t)m * HSZ + ncol;
      const float cn = fg * c[cidx] + ig * gg;
      const float hn = og * tanhf(cn);
      c[cidx] = cn;
      hOut[cidx] = (__bf16)hn;
      const int tr = (m < BATCH) ? s : (TLEN - 1 - s);
      const size_t xidx = ((size_t)tr * NB + m) * HSZ + ncol;
      Xout[xidx] = (__bf16)((float)Xin[xidx] + hn);  // residual (D==H, every layer)
      if (lastStep) {
        Hfin[cidx] = (__bf16)hn;
        Cfin[cidx] = (__bf16)cn;
      }
    }
  }
}

// ---------------- output projection: [B*T, 2H] @ Wo^T + bo ----------------
__global__ void __launch_bounds__(256)
proj_out_kernel(const __bf16* __restrict__ Xfin, const __bf16* __restrict__ WoB,
                const float* __restrict__ bo, float* __restrict__ out) {
  const int lane = threadIdx.x & 31;
  const int w    = threadIdx.x >> 5;
  const int gw = blockIdx.x * 8 + w;      // 16384 waves
  const int mT = gw & 1023;               // 1024 tiles of 16 rows
  const int nT = gw >> 10;                // 16 groups of 64 cols
  const int mtile = mT * 16;
  const int rsel  = lane & 15;

  const int r = mtile + rsel;             // row = t*64 + b
  const int t = r >> 6, b = r & 63;
  const __bf16* aF = Xfin + ((size_t)t * NB + b) * HSZ;
  const __bf16* aR = aF + (size_t)BATCH * HSZ;

  const __bf16* bP[4];
#pragma unroll
  for (int q = 0; q < 4; ++q)
    bP[q] = WoB + (size_t)(nT * 64 + q * 16 + rsel) * KCAT;

  v8f acc[4] = {v8f{}, v8f{}, v8f{}, v8f{}};
  for (int k0 = 0; k0 < KCAT; k0 += 32) {
    const v16bf a = (k0 < HSZ) ? load_frag(aF, k0, lane) : load_frag(aR, k0 - HSZ, lane);
#pragma unroll
    for (int q = 0; q < 4; ++q) {
      const v16bf bfr = load_frag(bP[q], k0, lane);
      acc[q] = __builtin_amdgcn_wmma_f32_16x16x32_bf16(false, a, false, bfr, (short)0, acc[q],
                                                       false, false);
    }
  }
  const int mbase = mtile + ((lane & 16) ? 8 : 0);
#pragma unroll
  for (int q = 0; q < 4; ++q) {
    const int n = nT * 64 + q * 16 + rsel;
    const float bias = bo[n];
#pragma unroll
    for (int vj = 0; vj < 8; ++vj) {
      const int rr = mbase + vj;
      const int tt = rr >> 6, bb = rr & 63;
      out[((size_t)bb * TLEN + tt) * HSZ + n] = acc[q][vj] + bias;
    }
  }
}

// ---------------- hidden/cell projection: [L*B, 2H] @ W^T + b ----------------
__global__ void __launch_bounds__(256)
proj_state_kernel(const __bf16* __restrict__ S, const __bf16* __restrict__ Wb,
                  const float* __restrict__ bias, float* __restrict__ out) {
  const int lane = threadIdx.x & 31;
  const int w    = threadIdx.x >> 5;
  const int gw = blockIdx.x * 8 + w;      // 256 waves
  const int mT = gw & 15;                 // 16 tiles of 16 rows (L*B=256)
  const int nT = gw >> 4;                 // 16 groups of 64 cols
  const int mtile = mT * 16;
  const int rsel  = lane & 15;

  const int r = mtile + rsel;             // row = l*64 + b
  const int l = r >> 6, b = r & 63;
  const __bf16* aF = S + ((size_t)l * NB + b) * HSZ;
  const __bf16* aR = aF + (size_t)BATCH * HSZ;

  const __bf16* bP[4];
#pragma unroll
  for (int q = 0; q < 4; ++q)
    bP[q] = Wb + (size_t)(nT * 64 + q * 16 + rsel) * KCAT;

  v8f acc[4] = {v8f{}, v8f{}, v8f{}, v8f{}};
  for (int k0 = 0; k0 < KCAT; k0 += 32) {
    const v16bf a = (k0 < HSZ) ? load_frag(aF, k0, lane) : load_frag(aR, k0 - HSZ, lane);
#pragma unroll
    for (int q = 0; q < 4; ++q) {
      const v16bf bfr = load_frag(bP[q], k0, lane);
      acc[q] = __builtin_amdgcn_wmma_f32_16x16x32_bf16(false, a, false, bfr, (short)0, acc[q],
                                                       false, false);
    }
  }
  const int mbase = mtile + ((lane & 16) ? 8 : 0);
#pragma unroll
  for (int q = 0; q < 4; ++q) {
    const int n = nT * 64 + q * 16 + rsel;
    const float bv = bias[n];
#pragma unroll
    for (int vj = 0; vj < 8; ++vj) {
      const int rr = mbase + vj;
      out[(size_t)rr * HSZ + n] = acc[q][vj] + bv;
    }
  }
}

// ---------------- host side ----------------
extern "C" void kernel_launch(void* const* d_in, const int* in_sizes, int n_in,
                              void* d_out, int out_size, void* d_ws, size_t ws_size,
                              hipStream_t stream) {
  (void)in_sizes; (void)n_in; (void)out_size; (void)ws_size;
  const int*   enc = (const int*)  d_in[0];
  const float* emb = (const float*)d_in[1];
  const float* Wih = (const float*)d_in[2];
  const float* Whh = (const float*)d_in[3];
  const float* bih = (const float*)d_in[4];
  const float* bhh = (const float*)d_in[5];
  const float* Wo  = (const float*)d_in[6];
  const float* bo  = (const float*)d_in[7];
  const float* Wh  = (const float*)d_in[8];
  const float* bh  = (const float*)d_in[9];
  const float* Wc  = (const float*)d_in[10];
  const float* bc  = (const float*)d_in[11];
  float* out = (float*)d_out;

  // workspace carve-up (256B aligned)
  char* ws = (char*)d_ws;
  size_t off = 0;
  auto carve = [&](size_t bytes) -> char* {
    char* p = ws + off;
    off = (off + bytes + 255) & ~(size_t)255;
    return p;
  };
  __bf16* WCAT = (__bf16*)carve((size_t)LNUM * G4H * KCAT * 2);  // 64 MB
  float*  BSUM = (float*) carve((size_t)LNUM * G4H * 4);
  __bf16* WOB  = (__bf16*)carve((size_t)HSZ * KCAT * 2);
  __bf16* WHB  = (__bf16*)carve((size_t)HSZ * KCAT * 2);
  __bf16* WCB  = (__bf16*)carve((size_t)HSZ * KCAT * 2);
  __bf16* X0   = (__bf16*)carve((size_t)TLEN * NB * HSZ * 2);    // 64 MB
  __bf16* X1   = (__bf16*)carve((size_t)TLEN * NB * HSZ * 2);    // 64 MB
  __bf16* HBF0 = (__bf16*)carve((size_t)NB * HSZ * 2);
  __bf16* HBF1 = (__bf16*)carve((size_t)NB * HSZ * 2);
  float*  CBUF = (float*) carve((size_t)NB * HSZ * 4);
  __bf16* HFIN = (__bf16*)carve((size_t)LNUM * NB * HSZ * 2);
  __bf16* CFIN = (__bf16*)carve((size_t)LNUM * NB * HSZ * 2);

  prep_wcat_kernel<<<4096, 256, 0, stream>>>(Wih, Whh, WCAT);
  prep_bsum_kernel<<<64, 256, 0, stream>>>(bih, bhh, BSUM);
  prep_proj_kernel<<<4096, 256, 0, stream>>>(Wo, Wh, Wc, WOB, WHB, WCB);
  embed_kernel<<<BATCH * TLEN, 256, 0, stream>>>(enc, emb, X0);

  __bf16* xin = X0;
  __bf16* xout = X1;
  for (int l = 0; l < LNUM; ++l) {
    zero_state_kernel<<<512, 256, 0, stream>>>(HBF0, CBUF);
    for (int s = 0; s < TLEN; ++s) {
      const __bf16* hin = (s & 1) ? HBF1 : HBF0;
      __bf16*       hout = (s & 1) ? HBF0 : HBF1;
      lstm_step_kernel<<<32, 256, 0, stream>>>(
          xin, xout, hin, hout, CBUF,
          WCAT + (size_t)l * G4H * KCAT, BSUM + l * G4H,
          HFIN + (size_t)l * NB * HSZ, CFIN + (size_t)l * NB * HSZ,
          s, (s == TLEN - 1) ? 1 : 0);
    }
    __bf16* tmp = xin; xin = xout; xout = tmp;   // layer output -> next input
  }

  // xin now holds final per-time outputs (fwd rows 0..63, bwd rows 64..127)
  proj_out_kernel<<<2048, 256, 0, stream>>>(xin, WOB, bo, out);
  const size_t OUT_H = (size_t)BATCH * TLEN * HSZ;                // 16,777,216
  const size_t ST_SZ = (size_t)LNUM * BATCH * HSZ;                // 262,144
  proj_state_kernel<<<32, 256, 0, stream>>>(HFIN, WHB, bh, out + OUT_H);
  proj_state_kernel<<<32, 256, 0, stream>>>(CFIN, WCB, bc, out + OUT_H + ST_SZ);
}
